// SimpleRNN_62886911148760
// MI455X (gfx1250) — matrix operand order repaired and checked by reading, not compiled, not run on hardware
//
#include <hip/hip_runtime.h>

// ---------------------------------------------------------------------------
// 2-layer Elman RNN for MI455X (gfx1250, wave32, WMMA).
// Pipeline:
//   1) pack_w_kernel : fp32 weights -> bf16 in WMMA B-operand layout
//   2) zproj_kernel  : Z[t,b,h] = x_t @ W_ih0^T + b_ih0 + b_hh0   (WMMA)
//   3) seq_kernel    : single persistent workgroup, 2048 recurrent steps;
//                      weights resident in LDS (256KB) + VGPRs, state in LDS
//                      in A-operand layout; Z software-pipelined in registers;
//                      split dual accumulation chains per layer.
//   4) expand_kernel : hs (16MB) -> y (512MB), non-temporal b128 stores
// ---------------------------------------------------------------------------

typedef __attribute__((ext_vector_type(16))) __bf16 v16bf;
typedef __attribute__((ext_vector_type(8)))  float  v8f;
typedef __attribute__((ext_vector_type(4)))  float  v4f;

#define T_LEN 2048
#define B_SZ  8
#define I_SZ  128
#define H_SZ  256
#define OUT_SZ 16

// WMMA 16-bit A-operand element mapping: lane holds row m=lane&15;
// element j of the v16bf maps to k-local:
__device__ __forceinline__ int kA_map(int lane, int j) {
  return ((lane >> 4) & 1) * 8 + (j & 7) + ((j >> 3) << 4);
}

// ---------------------------------------------------------------------------
// Pack row-major fp32 W[N][K] into bf16 WMMA B-operand tiles:
// dst[((nt*KC + kc)*32 + lane)*16 + j] = W[nt*16 + (lane&15)][kc*32 + kB]
// with kB = kA_map(lane, j) = 8*lane[4] + (j&7) + 16*(j>>3)  (B-matrix 32x16 split layout, n = lane&15).
// ---------------------------------------------------------------------------
__global__ void pack_w_kernel(const float* __restrict__ W, __bf16* __restrict__ dst,
                              int K, int total) {
  int idx = blockIdx.x * blockDim.x + threadIdx.x;
  if (idx >= total) return;
  int j    = idx & 15;
  int lane = (idx >> 4) & 31;
  int rest = idx >> 9;          // nt*KC + kc
  int KC   = K >> 5;
  int kc   = rest % KC;
  int nt   = rest / KC;
  int n    = nt * 16 + (lane & 15);
  int k    = kc * 32 + kA_map(lane, j);
  dst[idx] = (__bf16)W[n * K + k];
}

// ---------------------------------------------------------------------------
// Z[t][b][n] = sum_i x[b][i][t] * W_ih0[n][i] + b_ih0[n] + b_hh0[n]
// One block per t, 16 waves, wave w owns columns [16w,16w+16).
// ---------------------------------------------------------------------------
__global__ __launch_bounds__(512) void zproj_kernel(
    const float* __restrict__ x, const __bf16* __restrict__ Wih0p,
    const float* __restrict__ b_ih0, const float* __restrict__ b_hh0,
    float* __restrict__ Z) {
  __shared__ __attribute__((aligned(32))) __bf16 xA[4 * 32 * 16];  // 4KB, A-layout

  int t   = blockIdx.x;
  int tid = threadIdx.x;

  // Stage x_t into A-operand layout (pad batch rows 8..15 with zeros).
  {
    int j = tid & 15, ln = (tid >> 4) & 31, kc = tid >> 9;
    int m = ln & 15;
    int k = kc * 32 + kA_map(ln, j);
    float v = (m < B_SZ) ? x[(m * I_SZ + k) * T_LEN + t] : 0.0f;
    xA[tid] = (__bf16)v;
  }
  __syncthreads();

  int w = tid >> 5, lane = tid & 31;
  int n = w * 16 + (lane & 15);
  float bias = b_ih0[n] + b_hh0[n];

  v8f acc;
#pragma unroll
  for (int r = 0; r < 8; ++r) acc[r] = bias;

#pragma unroll
  for (int kc = 0; kc < 4; ++kc) {   // K = 128 = 4 * 32
    v16bf a = *(const v16bf*)&xA[(kc * 32 + lane) * 16];
    v16bf b = *(const v16bf*)&Wih0p[((w * 4 + kc) * 32 + lane) * 16];
    acc = __builtin_amdgcn_wmma_f32_16x16x32_bf16(false, a, false, b,
                                                  (short)0, acc, false, false);
  }

  if (lane < 16) {
#pragma unroll
    for (int r = 0; r < 8; ++r)
      Z[((size_t)t * B_SZ + r) * H_SZ + n] = acc[r];
  }
}

// ---------------------------------------------------------------------------
// Sequential recurrence: one workgroup, 16 waves, 2048 steps.
// LDS: hA0/hA1 ping-pong state (A-layout), W_hh0 + W_ih1 packed bf16
// (128KB each). W_hh1 lives in VGPRs. Z is register-pipelined one step ahead.
// ---------------------------------------------------------------------------
__global__ __launch_bounds__(512) void seq_kernel(
    const float* __restrict__ Z, const float* __restrict__ h0in,
    const __bf16* __restrict__ Whh0p, const __bf16* __restrict__ Wih1p,
    const __bf16* __restrict__ Whh1p,
    const float* __restrict__ b_ih1, const float* __restrict__ b_hh1,
    float* __restrict__ hs) {
  extern __shared__ __attribute__((aligned(32))) char smem[];
  __bf16* hA0  = (__bf16*)smem;        // [2][4096]
  __bf16* hA1  = hA0 + 2 * 4096;       // [2][4096]
  __bf16* Wh0s = hA1 + 2 * 4096;       // [65536]  (128 KB)
  __bf16* Wi1s = Wh0s + 65536;         // [65536]  (128 KB)

  int tid  = threadIdx.x;
  int w    = tid >> 5;
  int lane = tid & 31;
  int lane15 = lane & 15;

  // Stage both LDS-resident weight matrices (128KB each) with 16B vectors.
  {
    const uint4* s0 = (const uint4*)Whh0p;
    const uint4* s1 = (const uint4*)Wih1p;
    uint4* d0 = (uint4*)Wh0s;
    uint4* d1 = (uint4*)Wi1s;
    for (int i = tid; i < 8192; i += 512) { d0[i] = s0[i]; d1[i] = s1[i]; }
  }

  // W_hh1 tiles for this wave -> registers (8 * v16bf = 64 VGPRs).
  v16bf wr[8];
#pragma unroll
  for (int kc = 0; kc < 8; ++kc)
    wr[kc] = *(const v16bf*)&Whh1p[((w * 8 + kc) * 32 + lane) * 16];

  // Initialize both layer states from h0 input, in A-operand layout.
  for (int idx = tid; idx < 4096; idx += 512) {
    int j = idx & 15, ln = (idx >> 4) & 31, kc = idx >> 9;
    int m = ln & 15;
    int k = kc * 32 + kA_map(ln, j);
    __bf16 v = (__bf16)((m < B_SZ) ? h0in[m * H_SZ + k] : 0.0f);
    hA0[idx] = v;
    hA1[idx] = v;
  }

  int n = w * 16 + lane15;
  float bias1 = b_ih1[n] + b_hh1[n];

  // Scatter constants: writing value (m=r, hidden n) into A-layout.
  // n = 16w + lane15 -> chunk = w>>1, kl = ((w&1)<<4)|lane15
  int sc_base = (((w >> 1) * 32 + ((lane15 >> 3) << 4)) * 16) +
                (((w & 1) << 3) | (lane15 & 7));      // + r*16 per row

  // Software-pipelined Z tile: zreg holds Z[t] while loads for Z[t+1] fly.
  float zreg[8];
#pragma unroll
  for (int r = 0; r < 8; ++r) zreg[r] = 0.0f;
  if (lane < 16) {
#pragma unroll
    for (int r = 0; r < 8; ++r) zreg[r] = Z[(size_t)r * H_SZ + n];
  }

  __syncthreads();

  for (int t = 0; t < T_LEN; ++t) {
    int cur = (t & 1) * 4096;
    int nxt = cur ^ 4096;

    // ---- layer 0: dual accumulation chains seeded with Z[t] (biases folded)
    v8f accA, accB = {};
#pragma unroll
    for (int r = 0; r < 8; ++r) accA[r] = zreg[r];

    // Kick off next step's Z loads immediately; latency hides behind WMMAs.
    if (t + 1 < T_LEN && lane < 16) {
      const float* zrow = Z + (size_t)(t + 1) * B_SZ * H_SZ + n;
#pragma unroll
      for (int r = 0; r < 8; ++r) zreg[r] = zrow[r * H_SZ];
    }

#pragma unroll
    for (int kc = 0; kc < 8; ++kc) {
      v16bf a = *(const v16bf*)&hA0[cur + (kc * 32 + lane) * 16];
      v16bf b = *(const v16bf*)&Wh0s[((w * 8 + kc) * 32 + lane) * 16];
      if (kc & 1)
        accB = __builtin_amdgcn_wmma_f32_16x16x32_bf16(false, a, false, b,
                                                       (short)0, accB, false, false);
      else
        accA = __builtin_amdgcn_wmma_f32_16x16x32_bf16(false, a, false, b,
                                                       (short)0, accA, false, false);
    }
    if (lane < 16) {
#pragma unroll
      for (int r = 0; r < 8; ++r)
        hA0[nxt + sc_base + r * 16] = (__bf16)tanhf(accA[r] + accB[r]);
    }
    __syncthreads();  // h0_new visible to all waves

    // ---- layer 1: two independent chains (Wih1 from LDS, Whh1 from VGPRs)
    v8f acc1, acc2 = {};
#pragma unroll
    for (int r = 0; r < 8; ++r) acc1[r] = bias1;
#pragma unroll
    for (int kc = 0; kc < 8; ++kc) {
      v16bf a0 = *(const v16bf*)&hA0[nxt + (kc * 32 + lane) * 16];
      v16bf b0 = *(const v16bf*)&Wi1s[((w * 8 + kc) * 32 + lane) * 16];
      acc1 = __builtin_amdgcn_wmma_f32_16x16x32_bf16(false, a0, false, b0,
                                                     (short)0, acc1, false, false);
      v16bf a1 = *(const v16bf*)&hA1[cur + (kc * 32 + lane) * 16];
      acc2 = __builtin_amdgcn_wmma_f32_16x16x32_bf16(false, a1, false, wr[kc],
                                                     (short)0, acc2, false, false);
    }
    if (lane < 16) {
      float* hrow = hs + (size_t)t * B_SZ * H_SZ + n;
#pragma unroll
      for (int r = 0; r < 8; ++r) {
        float hv = tanhf(acc1[r] + acc2[r]);
        hA1[nxt + sc_base + r * 16] = (__bf16)hv;
        hrow[r * H_SZ] = hv;
      }
    }
    __syncthreads();  // h1_new visible, old buffers free for reuse
  }
}

// ---------------------------------------------------------------------------
// y[t][s][b][h][o] = hs[t][b][h] * W_fc[o] + b_fc[o]
// 512 MB (2.7x L2): pure streaming writes -> non-temporal b128 stores.
// ---------------------------------------------------------------------------
__global__ void expand_kernel(const float* __restrict__ hs,
                              const float* __restrict__ Wfc,
                              const float* __restrict__ bfc,
                              float* __restrict__ y) {
  int idx = blockIdx.x * blockDim.x + threadIdx.x;   // one per (t,b,h)
  if (idx >= T_LEN * B_SZ * H_SZ) return;
  float v = hs[idx];
  v4f o[4];
#pragma unroll
  for (int q = 0; q < 4; ++q) {
    o[q].x = v * Wfc[q * 4 + 0] + bfc[q * 4 + 0];
    o[q].y = v * Wfc[q * 4 + 1] + bfc[q * 4 + 1];
    o[q].z = v * Wfc[q * 4 + 2] + bfc[q * 4 + 2];
    o[q].w = v * Wfc[q * 4 + 3] + bfc[q * 4 + 3];
  }
  int t  = idx >> 11;          // B*H = 2048
  int bh = idx & 2047;
  v4f* d0 = (v4f*)(y + (((size_t)t * 2 + 0) * 2048 + bh) * OUT_SZ);
  v4f* d1 = (v4f*)(y + (((size_t)t * 2 + 1) * 2048 + bh) * OUT_SZ);
#pragma unroll
  for (int q = 0; q < 4; ++q) {
    __builtin_nontemporal_store(o[q], &d0[q]);
    __builtin_nontemporal_store(o[q], &d1[q]);
  }
}

// ---------------------------------------------------------------------------
extern "C" void kernel_launch(void* const* d_in, const int* in_sizes, int n_in,
                              void* d_out, int out_size, void* d_ws, size_t ws_size,
                              hipStream_t stream) {
  (void)in_sizes; (void)n_in; (void)out_size; (void)ws_size;

  const float* h0    = (const float*)d_in[0];
  // d_in[1] = c0 (unused by the cell)
  const float* x     = (const float*)d_in[2];
  const float* W_ih0 = (const float*)d_in[3];
  const float* b_ih0 = (const float*)d_in[4];
  const float* W_hh0 = (const float*)d_in[5];
  const float* b_hh0 = (const float*)d_in[6];
  const float* W_ih1 = (const float*)d_in[7];
  const float* b_ih1 = (const float*)d_in[8];
  const float* W_hh1 = (const float*)d_in[9];
  const float* b_hh1 = (const float*)d_in[10];
  const float* W_fc  = (const float*)d_in[11];
  const float* b_fc  = (const float*)d_in[12];

  const size_t TBH = (size_t)T_LEN * B_SZ * H_SZ;   // 4,194,304
  float*  Z     = (float*)d_ws;            // 16 MB
  float*  hs    = Z + TBH;                 // 16 MB
  __bf16* Wih0p = (__bf16*)(hs + TBH);     // 64 KB
  __bf16* Whh0p = Wih0p + 32768;           // 128 KB
  __bf16* Wih1p = Whh0p + 65536;           // 128 KB
  __bf16* Whh1p = Wih1p + 65536;           // 128 KB

  pack_w_kernel<<<128, 256, 0, stream>>>(W_ih0, Wih0p, I_SZ, 32768);
  pack_w_kernel<<<256, 256, 0, stream>>>(W_hh0, Whh0p, H_SZ, 65536);
  pack_w_kernel<<<256, 256, 0, stream>>>(W_ih1, Wih1p, H_SZ, 65536);
  pack_w_kernel<<<256, 256, 0, stream>>>(W_hh1, Whh1p, H_SZ, 65536);

  zproj_kernel<<<T_LEN, 512, 0, stream>>>(x, Wih0p, b_ih0, b_hh0, Z);

  size_t seq_lds = (size_t)(2 * 4096 + 2 * 4096 + 65536 + 65536) * sizeof(__bf16); // 288 KB
  seq_kernel<<<1, 512, seq_lds, stream>>>(Z, h0, Whh0p, Wih1p, Whh1p,
                                          b_ih1, b_hh1, hs);

  expand_kernel<<<(int)((TBH + 255) / 256), 256, 0, stream>>>(hs, W_fc, b_fc,
                                                              (float*)d_out);
}
